// LlamaAttention_heavy_hitter_37245956391398
// MI455X (gfx1250) — matrix operand order repaired and checked by reading, not compiled, not run on hardware
//
#include <hip/hip_runtime.h>
#include <hip/hip_bf16.h>

// ---------------------------------------------------------------------------
// LlamaAttention + H2O heavy-hitter pruning for MI455X (gfx1250, wave32).
// All GEMMs run on v_wmma_f32_16x16x32_bf16 (f32 accumulate). Working set
// (weights 256MB f32) is L2-resident on the 192MB L2; fragments are loaded
// directly from global with on-the-fly f32->bf16 conversion.
// ---------------------------------------------------------------------------

#define S_DIM 1024
#define D_DIM 4096
#define H_NUM 32
#define HD_DIM 128
#define HB 102          // heavy budget = int(0.1*1024)
#define RB 102          // recent budget
#define NEG_F (-3.4028234663852886e38f)

typedef __attribute__((ext_vector_type(16))) __bf16 v16bf;
typedef __attribute__((ext_vector_type(8)))  float  v8f;

// ---------------------------------------------------------------------------
// Tiled WMMA GEMM.
//   BT = true  : C[M,N] = scale * A[M,K] * B[N,K]^T   (A,B row-major, K contig)
//   BT = false : C[M,N] = scale * A[M,K] * B[K,N]     (B row-major over N)
// Block = 256 threads = 8 waves; wave computes a 32x32 tile (2x2 WMMA tiles);
// block tile = 128(M) x 64(N). All dims here are multiples of the tiles.
// Fragment layouts follow CDNA5 ISA 7.12.2 (wave32).
// ---------------------------------------------------------------------------
template <bool BT>
__global__ __launch_bounds__(256) void gemm_wmma_bf16(
    const float* __restrict__ A, const float* __restrict__ B,
    float* __restrict__ C, int M, int N, int K,
    int lda, int ldb, int ldc,
    long strideAz, long strideBz, long strideCz, float scale)
{
    const int z = blockIdx.z;
    A += (long)z * strideAz;
    B += (long)z * strideBz;
    C += (long)z * strideCz;

    const int tid   = threadIdx.x;
    const int wave  = tid >> 5;        // 0..7
    const int lane  = tid & 31;
    const int waveM = wave >> 1;       // 0..3
    const int waveN = wave & 1;        // 0..1
    const int m0 = blockIdx.y * 128 + waveM * 32;
    const int n0 = blockIdx.x * 64  + waveN * 32;
    const int lm = lane & 15;          // row/col within 16
    const int lh = lane >> 4;          // lane half: 0 or 1

    v8f c00 = {}, c01 = {}, c10 = {}, c11 = {};

    for (int kb = 0; kb < K; kb += 32) {
        // ---- A fragments: 16x32 bf16, lane lm = M row, lh selects K halves.
        // elements 0..7  -> K = kb + lh*8 + e
        // elements 8..15 -> K = kb + 16 + lh*8 + (e-8)
        v16bf a0, a1;
        {
            const float* ar0 = A + (long)(m0 + lm)      * lda + kb;
            const float* ar1 = A + (long)(m0 + 16 + lm) * lda + kb;
#pragma unroll
            for (int e = 0; e < 8; ++e) {
                a0[e]     = (__bf16)ar0[lh * 8 + e];
                a0[e + 8] = (__bf16)ar0[16 + lh * 8 + e];
                a1[e]     = (__bf16)ar1[lh * 8 + e];
                a1[e + 8] = (__bf16)ar1[16 + lh * 8 + e];
            }
        }
        // ---- B fragments: 32x16 bf16, lane lm = N col, same K order as A
        v16bf b0, b1;
        {
            const int nn0 = n0 + lm;
            const int nn1 = n0 + 16 + lm;
#pragma unroll
            for (int e = 0; e < 16; ++e) {
                const int kk = kb + ((e < 8) ? (lh * 8 + e) : (16 + lh * 8 + (e - 8)));
                if (BT) {
                    b0[e] = (__bf16)B[(long)nn0 * ldb + kk];
                    b1[e] = (__bf16)B[(long)nn1 * ldb + kk];
                } else {
                    b0[e] = (__bf16)B[(long)kk * ldb + nn0];
                    b1[e] = (__bf16)B[(long)kk * ldb + nn1];
                }
            }
        }
        c00 = __builtin_amdgcn_wmma_f32_16x16x32_bf16(false, a0, false, b0, (short)0, c00, false, false);
        c01 = __builtin_amdgcn_wmma_f32_16x16x32_bf16(false, a0, false, b1, (short)0, c01, false, false);
        c10 = __builtin_amdgcn_wmma_f32_16x16x32_bf16(false, a1, false, b0, (short)0, c10, false, false);
        c11 = __builtin_amdgcn_wmma_f32_16x16x32_bf16(false, a1, false, b1, (short)0, c11, false, false);
    }

    // C/D layout: VGPR r -> M = r + 8*lh ; lane lm -> N.
#pragma unroll
    for (int r = 0; r < 8; ++r) {
        const int mr0 = m0 + r + 8 * lh;
        const int mr1 = m0 + 16 + r + 8 * lh;
        const int nc0 = n0 + lm;
        const int nc1 = n0 + 16 + lm;
        C[(long)mr0 * ldc + nc0] = c00[r] * scale;
        C[(long)mr0 * ldc + nc1] = c01[r] * scale;
        C[(long)mr1 * ldc + nc0] = c10[r] * scale;
        C[(long)mr1 * ldc + nc1] = c11[r] * scale;
    }
}

// ---------------------------------------------------------------------------
// RoPE applied in place to Q and K, both [S, D] with head-interleaved columns.
// One thread per (s, h, j<64): rotates the (j, j+64) pair.
// ---------------------------------------------------------------------------
__global__ __launch_bounds__(256) void rope_kernel(float* __restrict__ Q,
                                                   float* __restrict__ Kp)
{
    const int idx = blockIdx.x * blockDim.x + threadIdx.x;
    if (idx >= S_DIM * H_NUM * 64) return;
    const int j = idx & 63;
    const int h = (idx >> 6) & (H_NUM - 1);
    const int s = idx >> 11;
    // inv_freq = 10000^(-2j/128)
    const float inv = __expf(-((float)(2 * j) / (float)HD_DIM) * 9.210340371976184f);
    const float ang = (float)s * inv;
    const float c = __cosf(ang), sn = __sinf(ang);
    const long o = (long)s * D_DIM + h * HD_DIM + j;
    const float q1 = Q[o], q2 = Q[o + 64];
    Q[o]      = q1 * c - q2 * sn;
    Q[o + 64] = q2 * c + q1 * sn;
    const float k1 = Kp[o], k2 = Kp[o + 64];
    Kp[o]      = k1 * c - k2 * sn;
    Kp[o + 64] = k2 * c + k1 * sn;
}

// ---------------------------------------------------------------------------
// Block-wide reductions / scan helpers (1024 threads).
// ---------------------------------------------------------------------------
__device__ __forceinline__ float blockMaxF(float v, float* sh, int tid) {
    sh[tid] = v; __syncthreads();
    for (int off = 512; off > 0; off >>= 1) {
        if (tid < off) sh[tid] = fmaxf(sh[tid], sh[tid + off]);
        __syncthreads();
    }
    const float r = sh[0]; __syncthreads();
    return r;
}
__device__ __forceinline__ float blockSumF(float v, float* sh, int tid) {
    sh[tid] = v; __syncthreads();
    for (int off = 512; off > 0; off >>= 1) {
        if (tid < off) sh[tid] += sh[tid + off];
        __syncthreads();
    }
    const float r = sh[0]; __syncthreads();
    return r;
}
__device__ __forceinline__ int blockSumI(int v, int* sh, int tid) {
    sh[tid] = v; __syncthreads();
    for (int off = 512; off > 0; off >>= 1) {
        if (tid < off) sh[tid] += sh[tid + off];
        __syncthreads();
    }
    const int r = sh[0]; __syncthreads();
    return r;
}

// ---------------------------------------------------------------------------
// H2O heavy-hitter scan + final masked softmax, fused. One block per head,
// 1024 threads; thread tid owns column tid. Overwrites scores[h] rows with
// the final attention probabilities.
// ---------------------------------------------------------------------------
__global__ __launch_bounds__(1024) void h2o_kernel(float* __restrict__ scores)
{
    const int tid = threadIdx.x;
    float* base = scores + (long)blockIdx.x * S_DIM * S_DIM;

    __shared__ float fsh[S_DIM];
    __shared__ int   ish[S_DIM];

    float acc = 0.0f;                       // accumulated H2O score, column tid

    for (int t = 0; t < S_DIM; ++t) {
        const float raw = base[(long)t * S_DIM + tid];
        const float causal = (tid <= t) ? raw : NEG_F;

        // plain causal softmax prob for the scan
        const float mx = blockMaxF(causal, fsh, tid);
        const float ex = __expf(causal - mx);
        const float sm = blockSumF(ex, fsh, tid);
        const float p  = ex / sm;

        int m;
        if (t < HB) {
            acc += p;                       // acc0 accumulation; masked at t==HB
            m = (tid < HB) ? 1 : 0;         // init block mask
        } else {
            if (t == HB && tid >= HB) acc = 0.0f;  // acc0 *= (cols < HB)
            // --- exact top-(HB-1) of acc via radix select over float bits ---
            const unsigned key = __float_as_uint(acc);  // acc >= 0 -> monotone
            const int k = HB - 1;
            unsigned prefix = 0u;
            for (int bit = 31; bit >= 0; --bit) {
                const unsigned cand = prefix | (1u << bit);
                const int cnt = blockSumI((key >= cand) ? 1 : 0, ish, tid);
                if (cnt >= k) prefix = cand;
            }
            const unsigned kv = prefix;                 // k-th largest key
            const int gt = blockSumI((key > kv) ? 1 : 0, ish, tid);
            const int need = k - gt;                    // ties admitted, low idx first
            const int eq = (key == kv) ? 1 : 0;
            ish[tid] = eq; __syncthreads();
            for (int off = 1; off < S_DIM; off <<= 1) { // inclusive scan of ties
                const int tmp = (tid >= off) ? ish[tid - off] : 0;
                __syncthreads();
                ish[tid] += tmp;
                __syncthreads();
            }
            const int selEq = eq && (ish[tid] <= need);
            __syncthreads();
            m = (key > kv) || selEq || (tid < 4) || (tid == t);
            acc = (acc + p) * (m ? 1.0f : 0.0f);
        }

        // final mask: (m | recent) & causal ; re-softmax ; store prob in place
        const int recent = (tid >= t - RB) ? 1 : 0;
        const int keep = ((m || recent) && (tid <= t)) ? 1 : 0;
        const float a = keep ? causal : NEG_F;
        const float mx2 = blockMaxF(a, fsh, tid);
        const float ex2 = __expf(a - mx2);
        const float sm2 = blockSumF(ex2, fsh, tid);
        base[(long)t * S_DIM + tid] = ex2 / sm2;
        __syncthreads();
    }
}

// ---------------------------------------------------------------------------
// Launch. Inputs: [hidden, attention_mask(unused; causal regenerated), Wq, Wk,
// Wv, Wo], all f32. Output f32 [1, S, D]. Workspace layout (floats):
//   Qf [S*D] | Kf [S*D] | Vf [S*D] | scores [H*S*S]    (attnOut aliases Qf)
// = 16 + 16 + 16 + 128 = 176 MB.
// ---------------------------------------------------------------------------
extern "C" void kernel_launch(void* const* d_in, const int* in_sizes, int n_in,
                              void* d_out, int out_size, void* d_ws, size_t ws_size,
                              hipStream_t stream) {
    const float* X  = (const float*)d_in[0];
    const float* Wq = (const float*)d_in[2];
    const float* Wk = (const float*)d_in[3];
    const float* Wv = (const float*)d_in[4];
    const float* Wo = (const float*)d_in[5];
    float* out = (float*)d_out;

    float* Qf = (float*)d_ws;
    float* Kf = Qf + (size_t)S_DIM * D_DIM;
    float* Vf = Kf + (size_t)S_DIM * D_DIM;
    float* SC = Vf + (size_t)S_DIM * D_DIM;   // [H][S][S]
    float* AO = Qf;                            // attnOut aliases Qf (Q dead)

    const dim3 blk(256);
    const float inv_sqrt_hd = 0.08838834764831845f;   // 1/sqrt(128)

    // 1) Q/K/V projections: C = X * W^T  (M=1024, N=4096, K=4096)
    {
        dim3 g(D_DIM / 64, S_DIM / 128, 1);
        gemm_wmma_bf16<true><<<g, blk, 0, stream>>>(X, Wq, Qf, S_DIM, D_DIM, D_DIM,
                                                    D_DIM, D_DIM, D_DIM, 0, 0, 0, 1.0f);
        gemm_wmma_bf16<true><<<g, blk, 0, stream>>>(X, Wk, Kf, S_DIM, D_DIM, D_DIM,
                                                    D_DIM, D_DIM, D_DIM, 0, 0, 0, 1.0f);
        gemm_wmma_bf16<true><<<g, blk, 0, stream>>>(X, Wv, Vf, S_DIM, D_DIM, D_DIM,
                                                    D_DIM, D_DIM, D_DIM, 0, 0, 0, 1.0f);
    }

    // 2) RoPE in place on Q, K
    {
        const int n = S_DIM * H_NUM * 64;
        rope_kernel<<<(n + 255) / 256, 256, 0, stream>>>(Qf, Kf);
    }

    // 3) scores[h] = (1/sqrt(hd)) * Q_h * K_h^T, batched over heads via z
    {
        dim3 g(S_DIM / 64, S_DIM / 128, H_NUM);
        gemm_wmma_bf16<true><<<g, blk, 0, stream>>>(
            Qf, Kf, SC, S_DIM, S_DIM, HD_DIM,
            D_DIM, D_DIM, S_DIM,
            (long)HD_DIM, (long)HD_DIM, (long)S_DIM * S_DIM, inv_sqrt_hd);
    }

    // 4) H2O heavy-hitter scan + final masked softmax (in place over SC)
    h2o_kernel<<<H_NUM, 1024, 0, stream>>>(SC);

    // 5) attnOut[:, h*128 : h*128+128] = P_h * V_h   (NN, batched over heads)
    {
        dim3 g(HD_DIM / 64, S_DIM / 128, H_NUM);
        gemm_wmma_bf16<false><<<g, blk, 0, stream>>>(
            SC, Vf, AO, S_DIM, HD_DIM, S_DIM,
            S_DIM, D_DIM, D_DIM,
            (long)S_DIM * S_DIM, (long)HD_DIM, (long)HD_DIM, 1.0f);
    }

    // 6) out = attnOut * Wo^T
    {
        dim3 g(D_DIM / 64, S_DIM / 128, 1);
        gemm_wmma_bf16<true><<<g, blk, 0, stream>>>(AO, Wo, out, S_DIM, D_DIM, D_DIM,
                                                    D_DIM, D_DIM, D_DIM, 0, 0, 0, 1.0f);
    }
}
